// SRNet_2585570312575
// MI455X (gfx1250) — matrix operand order repaired and checked by reading, not compiled, not run on hardware
//
#include <hip/hip_runtime.h>
#include <hip/hip_bf16.h>

typedef __attribute__((ext_vector_type(16))) _Float16 v16h;
typedef __attribute__((ext_vector_type(8)))  float    v8f;

constexpr int HP   = 255;          // H - 1
constexpr int WPQ  = 255;          // W - 1
constexpr int NPIX = 6 * HP * WPQ; // B*C*Hp*Wp = 390150
constexpr int STRD = 72;           // padded LDS row stride (halfs); 144B = 9*16B -> 16B aligned rows
constexpr int NFRAG = 94;          // total B fragments (see table in prep kernel)

// ---------------------------------------------------------------------------
// B-fragment load from global (pre-swizzled f16): each lane's 16 halfs are
// contiguous (32B) -> two b128 loads.
// ---------------------------------------------------------------------------
__device__ __forceinline__ v16h ld_bfrag(const _Float16* __restrict__ frags,
                                         int fragIdx, int lane) {
    union { uint4 u[2]; v16h h; } r;
    const uint4* p = (const uint4*)(frags + (size_t)fragIdx * 512 + lane * 16);
    r.u[0] = p[0];
    r.u[1] = p[1];
    return r.h;
}

// ---------------------------------------------------------------------------
// A-fragment load from LDS activation row (row = this lane's pixel m).
// A layout (16-bit, 16x32): slot j -> k = (j/8)*16 + (lane/16)*8 + (j%8),
// i.e. two 8-half contiguous runs -> two 16B LDS loads.
// ---------------------------------------------------------------------------
__device__ __forceinline__ v16h ld_afrag(const _Float16* row, int kb, int hi) {
    union { uint4 u[2]; v16h h; } r;
    r.u[0] = *(const uint4*)(row + kb + hi * 8);
    r.u[1] = *(const uint4*)(row + kb + 16 + hi * 8);
    return r.h;
}

// ---------------------------------------------------------------------------
// One dense layer: D(16x64) = A(16xK) * B(Kx64), bias+relu, store f16
// activations to this wave's LDS tile in [m][k] order for A-fragment reload.
// D layout: lane holds n = lane%16, rows m = r + 8*(lane/16).
// ---------------------------------------------------------------------------
template <int KSTEPS>
__device__ __forceinline__ void dense_layer(const _Float16* __restrict__ frags,
                                            int fragBase, const v16h* a,
                                            const float* __restrict__ bias,
                                            _Float16* lds, int lane) {
    const int n  = lane & 15;
    const int hi = lane >> 4;
#pragma unroll
    for (int t = 0; t < 4; ++t) {
        v8f acc = {};
#pragma unroll
        for (int s = 0; s < KSTEPS; ++s) {
            v16h bfr = ld_bfrag(frags, fragBase + t * KSTEPS + s, lane);
            acc = __builtin_amdgcn_wmma_f32_16x16x32_f16(
                false, a[s], false, bfr, (short)0, acc, false, false);
        }
        const float bv = bias[t * 16 + n];
#pragma unroll
        for (int r = 0; r < 8; ++r) {
            float v = fmaxf(acc[r] + bv, 0.0f);
            lds[(r + 8 * hi) * STRD + t * 16 + n] = (_Float16)v;
        }
    }
}

// ---------------------------------------------------------------------------
// Prep: swizzle f32 weights into f16 WMMA B-fragments in d_ws.
// Fragment element (lane, j): n = tile*16 + lane%16, k = step*32 + (j/8)*16 + (lane/16)*8 + (j%8)
// (CDNA5 16-bit B operand layout). Zero-pad k >= K, n >= N.
// Layout: [layer][tile][kstep], frag = 512 halfs.
//   L1: base 0  ks 1  (K=4 pad->32)   L2: base 4  ks 2  (K=64)
//   L3: base 12 ks 4  (K=128)         L4: base 28 ks 6  (K=192)
//   L5: base 52 ks 8  (K=256)         L6: base 84 ks 10 (K=320, N=4, 1 tile)
// ---------------------------------------------------------------------------
__global__ void srnet_prep_frags(const float* __restrict__ W1, const float* __restrict__ W2,
                                 const float* __restrict__ W3, const float* __restrict__ W4,
                                 const float* __restrict__ W5, const float* __restrict__ W6,
                                 _Float16* __restrict__ frags) {
    int e = blockIdx.x * 256 + threadIdx.x;
    if (e >= NFRAG * 512) return;
    int frag   = e >> 9;
    int within = e & 511;
    int lane   = within >> 4;
    int j      = within & 15;

    const float* W; int Kl, Nl, base, ks;
    if (frag < 4)       { W = W1; Kl = 4;   Nl = 64; base = 0;  ks = 1;  }
    else if (frag < 12) { W = W2; Kl = 64;  Nl = 64; base = 4;  ks = 2;  }
    else if (frag < 28) { W = W3; Kl = 128; Nl = 64; base = 12; ks = 4;  }
    else if (frag < 52) { W = W4; Kl = 192; Nl = 64; base = 28; ks = 6;  }
    else if (frag < 84) { W = W5; Kl = 256; Nl = 64; base = 52; ks = 8;  }
    else                { W = W6; Kl = 320; Nl = 4;  base = 84; ks = 10; }

    int f0 = frag - base;
    int t  = f0 / ks;
    int s  = f0 % ks;
    int n  = t * 16 + (lane & 15);
    int k  = s * 32 + (j >> 3) * 16 + (lane >> 4) * 8 + (j & 7);
    float v = (k < Kl && n < Nl) ? W[n * Kl + k] : 0.0f;
    frags[e] = (_Float16)v;
}

// ---------------------------------------------------------------------------
// Main: one wave = 16 pixels. Feature vector (<=320) persists in VGPRs as up
// to ten 16x32 f16 A-fragments; layers append fragments (free concat).
// ---------------------------------------------------------------------------
__global__ void __launch_bounds__(256) srnet_main(
    const float* __restrict__ x, const _Float16* __restrict__ frags,
    const float* __restrict__ b1, const float* __restrict__ b2,
    const float* __restrict__ b3, const float* __restrict__ b4,
    const float* __restrict__ b5, const float* __restrict__ b6,
    float* __restrict__ out) {
    __shared__ __align__(16) _Float16 smem[8][16 * STRD];

    const int lane   = threadIdx.x & 31;
    const int wv     = threadIdx.x >> 5;
    const int waveId = blockIdx.x * 8 + wv;
    const int base   = waveId * 16;
    if (base >= NPIX) return;          // whole-wave uniform early exit; no block barriers used

    const int n  = lane & 15;
    const int hi = lane >> 4;
    _Float16* lds = &smem[wv][0];
    const _Float16* row = lds + n * STRD;   // this lane's pixel row (m = lane%16)

    v16h afrag[10];

    // ---- layer 1: A = 2x2 patch, K=4 (padded to 32; lanes 16..31 and slots>=4 are zero)
    v16h a0 = {};
    {
        int p  = base + n;
        int pc = p < NPIX ? p : NPIX - 1;
        int bc  = pc / (HP * WPQ);
        int rem = pc - bc * (HP * WPQ);
        int ii  = rem / WPQ;
        int jj  = rem - ii * WPQ;
        const float* xp = x + bc * (256 * 256) + ii * 256 + jj;
        if (hi == 0) {
            a0[0] = (_Float16)xp[0];
            a0[1] = (_Float16)xp[1];
            a0[2] = (_Float16)xp[256];
            a0[3] = (_Float16)xp[257];
        }
    }
    dense_layer<1>(frags, 0, &a0, b1, lds, lane);
    asm volatile("s_wait_dscnt 0x0" ::: "memory");
    afrag[0] = ld_afrag(row, 0, hi);
    afrag[1] = ld_afrag(row, 32, hi);

    dense_layer<2>(frags, 4, afrag, b2, lds, lane);
    asm volatile("s_wait_dscnt 0x0" ::: "memory");
    afrag[2] = ld_afrag(row, 0, hi);
    afrag[3] = ld_afrag(row, 32, hi);

    dense_layer<4>(frags, 12, afrag, b3, lds, lane);
    asm volatile("s_wait_dscnt 0x0" ::: "memory");
    afrag[4] = ld_afrag(row, 0, hi);
    afrag[5] = ld_afrag(row, 32, hi);

    dense_layer<6>(frags, 28, afrag, b4, lds, lane);
    asm volatile("s_wait_dscnt 0x0" ::: "memory");
    afrag[6] = ld_afrag(row, 0, hi);
    afrag[7] = ld_afrag(row, 32, hi);

    dense_layer<8>(frags, 52, afrag, b5, lds, lane);
    asm volatile("s_wait_dscnt 0x0" ::: "memory");
    afrag[8] = ld_afrag(row, 0, hi);
    afrag[9] = ld_afrag(row, 32, hi);

    // ---- head: K=320 -> 4 outputs (single n-tile, n=0..3 valid), tanh,
    // pixel-shuffle scatter: channel n=(di*2+dj) -> out[2i+di][2j+dj]
    v8f acc = {};
#pragma unroll
    for (int s = 0; s < 10; ++s) {
        v16h bfr = ld_bfrag(frags, 84 + s, lane);
        acc = __builtin_amdgcn_wmma_f32_16x16x32_f16(
            false, afrag[s], false, bfr, (short)0, acc, false, false);
    }
    if (n < 4) {
        const float bv = b6[n];
        const int di = n >> 1, dj = n & 1;
#pragma unroll
        for (int r = 0; r < 8; ++r) {
            int p = base + r + 8 * hi;
            if (p < NPIX) {
                int bc  = p / (HP * WPQ);
                int rem = p - bc * (HP * WPQ);
                int ii  = rem / WPQ;
                int jj  = rem - ii * WPQ;
                float v = tanhf(acc[r] + bv);
                out[bc * (510 * 510) + (2 * ii + di) * 510 + (2 * jj + dj)] = v;
            }
        }
    }
}

extern "C" void kernel_launch(void* const* d_in, const int* in_sizes, int n_in,
                              void* d_out, int out_size, void* d_ws, size_t ws_size,
                              hipStream_t stream) {
    const float* x  = (const float*)d_in[0];
    const float* W1 = (const float*)d_in[1];
    const float* b1 = (const float*)d_in[2];
    const float* W2 = (const float*)d_in[3];
    const float* b2 = (const float*)d_in[4];
    const float* W3 = (const float*)d_in[5];
    const float* b3 = (const float*)d_in[6];
    const float* W4 = (const float*)d_in[7];
    const float* b4 = (const float*)d_in[8];
    const float* W5 = (const float*)d_in[9];
    const float* b5 = (const float*)d_in[10];
    const float* W6 = (const float*)d_in[11];
    const float* b6 = (const float*)d_in[12];
    float*    out   = (float*)d_out;
    _Float16* frags = (_Float16*)d_ws;   // needs 94*512*2 = 96,256 bytes

    srnet_prep_frags<<<(NFRAG * 512 + 255) / 256, 256, 0, stream>>>(
        W1, W2, W3, W4, W5, W6, frags);

    int nWaves = (NPIX + 15) / 16;      // 24385 waves of 16 pixels
    int blocks = (nWaves + 7) / 8;      // 8 waves (256 threads) per block
    srnet_main<<<blocks, 256, 0, stream>>>(x, frags, b1, b2, b3, b4, b5, b6, out);
}
